// MHAttention_own_52226802319926
// MI455X (gfx1250) — matrix operand order repaired and checked by reading, not compiled, not run on hardware
//
#include <hip/hip_runtime.h>
#include <hip/hip_bf16.h>
#include <stdint.h>

typedef __attribute__((ext_vector_type(16))) _Float16 v16h;
typedef __attribute__((ext_vector_type(8)))  float    v8f;
typedef unsigned int uint32x4 __attribute__((ext_vector_type(4)));
typedef int          int32x8  __attribute__((ext_vector_type(8)));
typedef int          int32x4  __attribute__((ext_vector_type(4)));

constexpr int S_LEN   = 1024;   // sequence length (32*32)
constexpr int D_HEAD  = 64;     // head dim (512 / 8 heads)
constexpr int Q_BLOCK = 128;    // query rows per block (16 per wave * 8 waves)
constexpr int K_CHUNK = 32;     // key columns processed per iteration
constexpr int M_STRIDE = 33;    // padded mask row stride in LDS (32 + 1 pad DWORD from TDM)

// ---- Tensor Data Mover: 2D f32 tile load global -> LDS (CDNA5 TDM, D# descriptor) ----
// Group0: count=1 | lds_addr | global_addr | type=2
// Group1: data_size=4B, pad_enable, pad_interval=32 DWORDs (code 4), pad_amount=1 DWORD (code 0)
__device__ __forceinline__ void tdm_load_2d_f32(unsigned ldsOff,
                                                unsigned long long gaddr,
                                                unsigned tile0, unsigned tile1,
                                                unsigned long long stride0)
{
    uint32x4 g0;
    g0[0] = 1u;                                               // count=1, user mode
    g0[1] = ldsOff;                                           // LDS byte address
    g0[2] = (unsigned)(gaddr & 0xFFFFFFFFull);                // global_addr[31:0]
    g0[3] = (unsigned)((gaddr >> 32) & 0x1FFFFFFull)          // global_addr[56:32]
          | 0x80000000u;                                      // type=2 ("image")

    const unsigned td0 = 1024, td1 = 1024;                    // tensor dims (tile always in-bounds)
    int32x8 g1;
    g1[0] = (int)((2u << 16) | (1u << 20) | (4u << 22));      // data_size=4B, pad_en, interval=32dw, amount=1dw
    g1[1] = (int)((td0 & 0xFFFFu) << 16);                     // atomic_barrier=0 | tensor_dim0 lo
    g1[2] = (int)((td0 >> 16) | ((td1 & 0xFFFFu) << 16));     // tensor_dim0 hi | tensor_dim1 lo
    g1[3] = (int)((td1 >> 16) | (tile0 << 16));               // tensor_dim1 hi | tile_dim0
    g1[4] = (int)(tile1 & 0xFFFFu);                           // tile_dim1 | tile_dim2=0
    g1[5] = (int)(unsigned)(stride0 & 0xFFFFFFFFull);         // tensor_dim0_stride lo
    g1[6] = (int)(unsigned)((stride0 >> 32) & 0xFFFFull);     // stride hi | dim1_stride lo = 0
    g1[7] = 0;

    int32x4 gz = {0, 0, 0, 0};
#if defined(__clang_major__) && __clang_major__ >= 23
    int32x8 gz8 = {0, 0, 0, 0, 0, 0, 0, 0};
    __builtin_amdgcn_tensor_load_to_lds(g0, g1, gz, gz, gz8, 0);
#else
    __builtin_amdgcn_tensor_load_to_lds(g0, g1, gz, gz, 0);
#endif
}

// Flash-attention forward, f16 WMMA with f32 accumulation.
//   A 16x32 f16 : lane L -> M = L%16, half = L/16; vgpr v -> K = (v/4)*16 + half*8 + (v%4)*2
//   B 32x16 f16 : lane L -> N = L%16, half = L/16; vgpr v -> K = (v/4)*16 + half*8 + (v%4)*2
//   C/D 16x16 f32: lane L -> N = L%16; vgpr r -> M = r + 8*(L/16)
__global__ __launch_bounds__(256, 1)
void fa_fwd_f16wmma(const float* __restrict__ qg, const float* __restrict__ kg,
                    const float* __restrict__ vg, const float* __restrict__ mg,
                    float* __restrict__ og)
{
    __shared__ _Float16 Kl[K_CHUNK][D_HEAD + 8];      // [tk][dd]
    __shared__ _Float16 Vl[D_HEAD][K_CHUNK + 8];      // [dd][tk]
    __shared__ _Float16 Pl[8][16][K_CHUNK + 8];       // per-wave P staging (C->A relayout)
    __shared__ float    Ml[2][128 * M_STRIDE];        // double-buffered TDM mask tiles

    const int tid  = threadIdx.x;
    const int wave = tid >> 5;
    const int lane = tid & 31;
    const int lmod = lane & 15;
    const int half = lane >> 4;

    const int bn   = blockIdx.y;          // 0..63 batch-head
    const int b    = bn >> 3;
    const int head = bn & 7;
    const int q0   = blockIdx.x * Q_BLOCK;

    const size_t base = ((size_t)b * 512 + (size_t)head * D_HEAD) * S_LEN;
    const unsigned long long mgaddr =
        (unsigned long long)(uintptr_t)mg +
        ((unsigned long long)b * S_LEN * S_LEN + (unsigned long long)q0 * S_LEN) * 4ull;

    const int tq = q0 + wave * 16 + lmod;             // A-matrix row of this lane

    // ---- Q fragments: two 16x32 f16 A-tiles covering d = 0..63 ----
    v16h qf[2];
    #pragma unroll
    for (int f = 0; f < 2; ++f) {
        #pragma unroll
        for (int vv = 0; vv < 8; ++vv) {
            int kk = ((vv >> 2) << 4) + (half << 3) + ((vv & 3) << 1);
            int dd = f * 32 + kk;
            qf[f][2 * vv + 0] = (_Float16)qg[base + (size_t)(dd + 0) * S_LEN + tq];
            qf[f][2 * vv + 1] = (_Float16)qg[base + (size_t)(dd + 1) * S_LEN + tq];
        }
    }

    // ---- prologue: TDM the first mask tile (128 x 32 f32, row stride 1024) ----
    if (wave == 0)
        tdm_load_2d_f32((unsigned)(uintptr_t)&Ml[0][0], mgaddr, K_CHUNK, Q_BLOCK, S_LEN);

    v8f oacc[4] = {};
    float mrow[8], lrow[8];
    #pragma unroll
    for (int r = 0; r < 8; ++r) { mrow[r] = -1e30f; lrow[r] = 0.0f; }

    const float scale = 0.125f;           // d^-0.5 = 1/8

    for (int kt0 = 0; kt0 < S_LEN; kt0 += K_CHUNK) {
        const int cur = (kt0 >> 5) & 1;
        const bool has_next = (kt0 + K_CHUNK) < S_LEN;

        __syncthreads();                  // all readers done with both LDS buffers

        // kick TDM for the NEXT mask tile into the other buffer (overlaps everything below)
        if (wave == 0 && has_next)
            tdm_load_2d_f32((unsigned)(uintptr_t)&Ml[cur ^ 1][0],
                            mgaddr + (unsigned long long)(kt0 + K_CHUNK) * 4ull,
                            K_CHUNK, Q_BLOCK, S_LEN);

        // ---- cooperative staging: K chunk (transposed) and V chunk into LDS (f16) ----
        {
            int tk  = tid & 31;
            int ddb = (tid >> 5) * 8;
            #pragma unroll
            for (int i = 0; i < 8; ++i)
                Kl[tk][ddb + i] = (_Float16)kg[base + (size_t)(ddb + i) * S_LEN + kt0 + tk];

            int dd = tid & 63;
            int cb = (tid >> 6) * 8;
            #pragma unroll
            for (int i = 0; i < 8; ++i)
                Vl[dd][cb + i] = (_Float16)vg[base + (size_t)dd * S_LEN + kt0 + cb + i];

            if (has_next) {
                __builtin_prefetch(&kg[base + (size_t)dd * S_LEN + kt0 + K_CHUNK], 0, 3);
                __builtin_prefetch(&vg[base + (size_t)dd * S_LEN + kt0 + K_CHUNK], 0, 3);
            }
        }

        // current mask tile's TDM (issued last iteration / prologue) must be done;
        // TDM completes in-order, so <=1 suffices while the next one is in flight.
        if (wave == 0) {
            if (has_next) __builtin_amdgcn_s_wait_tensorcnt(1);
            else          __builtin_amdgcn_s_wait_tensorcnt(0);
        }
        __syncthreads();

        // ---- S = Q * K^T for this wave's 16 rows x 32 keys ----
        v8f sacc[2];
        #pragma unroll
        for (int nt = 0; nt < 2; ++nt) {
            v8f acc = {};
            #pragma unroll
            for (int f = 0; f < 2; ++f) {
                v16h bf;
                #pragma unroll
                for (int vv = 0; vv < 8; ++vv) {
                    int kk = ((vv >> 2) << 4) + (half << 3) + ((vv & 3) << 1);
                    int dd = f * 32 + kk;
                    int n  = nt * 16 + lmod;
                    bf[2 * vv + 0] = Kl[n][dd + 0];
                    bf[2 * vv + 1] = Kl[n][dd + 1];
                }
                acc = __builtin_amdgcn_wmma_f32_16x16x32_f16(
                          false, qf[f], false, bf, (short)0, acc, false, false);
            }
            sacc[nt] = acc;
        }

        // ---- masked online softmax (mask from TDM-staged LDS tile) ----
        float sv[2][8], pm[2][8];
        #pragma unroll
        for (int nt = 0; nt < 2; ++nt) {
            int tkl = nt * 16 + lmod;
            #pragma unroll
            for (int r = 0; r < 8; ++r) {
                int rowl = wave * 16 + r + (half << 3);
                float mk  = Ml[cur][rowl * M_STRIDE + tkl];
                float kee = (mk > -5000.0f) ? 1.0f : 0.0f;    // mask==0 -> keep
                pm[nt][r] = kee;
                float x   = sacc[nt][r] * scale;
                sv[nt][r] = (kee != 0.0f) ? x : -1e30f;       // exclude from row max
            }
        }

        #pragma unroll
        for (int r = 0; r < 8; ++r) {
            float t = fmaxf(sv[0][r], sv[1][r]);
            #pragma unroll
            for (int off = 1; off < 16; off <<= 1)
                t = fmaxf(t, __shfl_xor(t, off, 32));

            float mnew = fmaxf(mrow[r], t);
            float esc  = __expf(mrow[r] - mnew);
            mrow[r] = mnew;

            float p0 = __expf(sv[0][r] - mnew) * pm[0][r];
            float p1 = __expf(sv[1][r] - mnew) * pm[1][r];
            sv[0][r] = p0;
            sv[1][r] = p1;

            float ps = p0 + p1;
            #pragma unroll
            for (int off = 1; off < 16; off <<= 1)
                ps += __shfl_xor(ps, off, 32);

            lrow[r] = lrow[r] * esc + ps;
            #pragma unroll
            for (int j = 0; j < 4; ++j)
                oacc[j][r] = oacc[j][r] * esc;
        }

        // ---- P: C layout -> LDS -> A layout ----
        #pragma unroll
        for (int nt = 0; nt < 2; ++nt)
            #pragma unroll
            for (int r = 0; r < 8; ++r)
                Pl[wave][r + (half << 3)][nt * 16 + lmod] = (_Float16)sv[nt][r];
        __syncthreads();

        v16h pf;
        #pragma unroll
        for (int vv = 0; vv < 8; ++vv) {
            int kk = ((vv >> 2) << 4) + (half << 3) + ((vv & 3) << 1);
            pf[2 * vv + 0] = Pl[wave][lmod][kk + 0];
            pf[2 * vv + 1] = Pl[wave][lmod][kk + 1];
        }

        // ---- O += P * V : four 16x16 d-tiles ----
        #pragma unroll
        for (int j = 0; j < 4; ++j) {
            v16h vf;
            #pragma unroll
            for (int vv = 0; vv < 8; ++vv) {
                int kk = ((vv >> 2) << 4) + (half << 3) + ((vv & 3) << 1);
                int dd = j * 16 + lmod;
                vf[2 * vv + 0] = Vl[dd][kk + 0];
                vf[2 * vv + 1] = Vl[dd][kk + 1];
            }
            oacc[j] = __builtin_amdgcn_wmma_f32_16x16x32_f16(
                          false, pf, false, vf, (short)0, oacc[j], false, false);
        }
    }

    // ---- epilogue: normalize and store (channel-major output) ----
    #pragma unroll
    for (int r = 0; r < 8; ++r) {
        float inv = 1.0f / lrow[r];
        int tqr = q0 + wave * 16 + r + (half << 3);
        #pragma unroll
        for (int j = 0; j < 4; ++j) {
            int dd = j * 16 + lmod;
            og[base + (size_t)dd * S_LEN + tqr] = oacc[j][r] * inv;
        }
    }
}

extern "C" void kernel_launch(void* const* d_in, const int* in_sizes, int n_in,
                              void* d_out, int out_size, void* d_ws, size_t ws_size,
                              hipStream_t stream) {
    const float* q    = (const float*)d_in[0];
    const float* k    = (const float*)d_in[1];
    const float* v    = (const float*)d_in[2];
    const float* mask = (const float*)d_in[3];
    float* out        = (float*)d_out;

    dim3 grid(S_LEN / Q_BLOCK, 64);   // 8 query tiles x 64 batch-heads
    fa_fwd_f16wmma<<<grid, 256, 0, stream>>>(q, k, v, mask, out);
}
